// MSEmbeddingNormNet_47021301957181
// MI455X (gfx1250) — matrix operand order repaired and checked by reading, not compiled, not run on hardware
//
#include <hip/hip_runtime.h>
#include <math.h>

#define BB 256
#define LL 512
#define F2 64      // 2F
#define HH 128
#define GG 512     // 4H
#define EPSV 1e-5f

typedef __attribute__((ext_vector_type(16))) _Float16 v16h;
typedef __attribute__((ext_vector_type(8)))  float    v8f;
typedef __attribute__((ext_vector_type(4)))  unsigned int v4u;

// ---- fast transcendentals on the hardware TRANS pipe (co-executes with WMMA) ----
__device__ __forceinline__ float fexp2(float x) {
#if __has_builtin(__builtin_amdgcn_exp2f)
  return __builtin_amdgcn_exp2f(x);   // v_exp_f32
#else
  return exp2f(x);
#endif
}
__device__ __forceinline__ float frcp(float x) {
#if __has_builtin(__builtin_amdgcn_rcpf)
  return __builtin_amdgcn_rcpf(x);    // v_rcp_f32
#else
  return 1.0f / x;
#endif
}
__device__ __forceinline__ float sigm(float x) {
  return frcp(1.0f + fexp2(x * -1.44269504088896340736f));
}
__device__ __forceinline__ float tanh_fast(float x) {
  const float xx = fminf(fmaxf(x, -16.0f), 16.0f);          // avoid inf*0 -> NaN
  const float e = fexp2(xx * -2.88539008177792681472f);     // exp(-2x)
  return fmaf(-2.0f * e, frcp(1.0f + e), 1.0f);             // (1-e)/(1+e)
}

// ---- WMMA fragment loaders (contiguous k = 16*(lane/16) + i in A and B, wave32) ----

// A (16x32 f16, MxK) from LDS, row-major with `stride` halfs per row.
__device__ __forceinline__ v16h load_a_frag(const _Float16* s, int stride, int k0) {
  const int lane = threadIdx.x & 31;
  const int m = lane & 15;
  const int kb = (lane >> 4) << 4;
  const _Float16* row = s + m * stride + k0 + kb;
  v16h a;
#pragma unroll
  for (int i = 0; i < 8; ++i) a[i] = row[i];          // ds_load_b128
#pragma unroll
  for (int i = 0; i < 8; ++i) a[8 + i] = row[8 + i]; // ds_load_b128
  return a;
}

// B (32x16 f16, KxN) from the fragment-packed GLOBAL buffer:
//   pk[((kblk*512 + gcol) << 5) + k_in_block], each lane reads 32 contiguous bytes.
__device__ __forceinline__ v16h load_b_frag_pk(const _Float16* __restrict__ pk,
                                               int kblk, int n0) {
  const int lane = threadIdx.x & 31;
  const int n = lane & 15;
  const int kb = (lane >> 4) << 4;
  const _Float16* p = pk + (((size_t)kblk * 512 + n0 + n) << 5) + kb;
  union { v4u u[2]; v16h h; } uh;
  uh.u[0] = *(const v4u*)p;        // global_load_b128
  uh.u[1] = *(const v4u*)(p + 8);  // global_load_b128
  return uh.h;
}

// xor-reduce across a 16-lane half-wave group (wave32)
__device__ __forceinline__ float red16(float v) {
  v += __shfl_xor(v, 1);
  v += __shfl_xor(v, 2);
  v += __shfl_xor(v, 4);
  v += __shfl_xor(v, 8);
  return v;
}

// =====================================================================
// Pack kernel: W (G=512 rows, K cols f32, K in {64,128}) -> fragment-packed f16
//   dst[((k>>5)*512 + g)*32 + (k&31)] = (f16)W[g*K + k]
// =====================================================================
__global__ void __launch_bounds__(256)
pack_kernel(const float* __restrict__ src, _Float16* __restrict__ dst, int kbits) {
  const int K = 1 << kbits;
  const int i = (blockIdx.x * 256 + threadIdx.x) * 2;
  const int g = i >> kbits;
  const int k = i & (K - 1);
  union { _Float16 h[2]; unsigned int u; } p;
  p.h[0] = (_Float16)src[i];
  p.h[1] = (_Float16)src[i + 1];
  *(unsigned int*)(dst + ((((size_t)(k >> 5) * 512) + g) << 5) + (k & 31)) = p.u;
}

// =====================================================================
// Phase 1: front-end MLPs + padding + x@W_ih^T (WMMA) + LN -> ig_all[t][b][512] f32
// grid = L * B/16; one WG = 16 rows (fixed t, 16 batch elems) x 512 cols
// =====================================================================
__global__ void __launch_bounds__(256)
ig_kernel(const float* __restrict__ peaks, const int* __restrict__ plen,
          const float* __restrict__ w_mz1, const float* __restrict__ b_mz1,
          const float* __restrict__ w_mz2, const float* __restrict__ b_mz2,
          const float* __restrict__ w_in1, const float* __restrict__ b_in1,
          const float* __restrict__ w_in2, const float* __restrict__ b_in2,
          const _Float16* __restrict__ pk_wih,
          const float* __restrict__ ln_i_w, const float* __restrict__ ln_i_b,
          float* __restrict__ ig) {
  extern __shared__ char smem[];
  _Float16* lds_x = (_Float16*)smem;            // 16x64 f16 = 2KB
  float* red_s  = (float*)(smem + 2048);        // 8 waves x 16 rows
  float* red_q  = red_s + 128;
  float* sm_mu  = red_q + 128;                  // 16
  float* sm_rs  = sm_mu + 16;                   // 16

  const int tid = threadIdx.x;
  const int t  = blockIdx.x >> 4;
  const int b0 = (blockIdx.x & 15) << 4;

  // front-end dual MLP: 16 threads/position, 2 output features each per channel
  {
    const int p  = tid >> 4;
    const int f0 = (tid & 15) << 1;
    const int b  = b0 + p;
    const float smz = peaks[((size_t)b * LL + t) * 2 + 0];
    const float sit = peaks[((size_t)b * LL + t) * 2 + 1];
    const bool pad = (t >= LL - plen[b]);   // reference pads the LAST peaksLen slots
    float a0 = b_mz2[f0], a1 = b_mz2[f0 + 1];
    float c0 = b_in2[f0], c1 = b_in2[f0 + 1];
#pragma unroll 8
    for (int k = 0; k < 32; ++k) {
      const float hm = fmaxf(fmaf(smz, w_mz1[k], b_mz1[k]), 0.0f);
      const float hi = fmaxf(fmaf(sit, w_in1[k], b_in1[k]), 0.0f);
      a0 = fmaf(hm, w_mz2[f0 * 32 + k], a0);
      a1 = fmaf(hm, w_mz2[(f0 + 1) * 32 + k], a1);
      c0 = fmaf(hi, w_in2[f0 * 32 + k], c0);
      c1 = fmaf(hi, w_in2[(f0 + 1) * 32 + k], c1);
    }
    a0 = pad ? 0.0f : fmaxf(a0, 0.0f);  a1 = pad ? 0.0f : fmaxf(a1, 0.0f);
    c0 = pad ? 0.0f : fmaxf(c0, 0.0f);  c1 = pad ? 0.0f : fmaxf(c1, 0.0f);
    lds_x[p * 64 + f0]          = (_Float16)a0;
    lds_x[p * 64 + f0 + 1]      = (_Float16)a1;
    lds_x[p * 64 + 32 + f0]     = (_Float16)c0;
    lds_x[p * 64 + 32 + f0 + 1] = (_Float16)c1;
  }
  __syncthreads();

  const int wv = tid >> 5, hw = (tid >> 4) & 1, n = tid & 15;

  // two independent accumulator chains (kk=0 / kk=1) -> no WMMA RAW hazard NOPs
  v8f acc[4], ac2[4];
#pragma unroll
  for (int tl = 0; tl < 4; ++tl)
#pragma unroll
    for (int r = 0; r < 8; ++r) { acc[tl][r] = 0.0f; ac2[tl][r] = 0.0f; }

  {
    const v16h a0f = load_a_frag(lds_x, 64, 0);
    const v16h a1f = load_a_frag(lds_x, 64, 32);
#pragma unroll
    for (int tl = 0; tl < 4; ++tl) {
      const v16h bb = load_b_frag_pk(pk_wih, 0, wv * 64 + tl * 16);
      acc[tl] = __builtin_amdgcn_wmma_f32_16x16x32_f16(
          false, a0f, false, bb, (short)0, acc[tl], false, false);
    }
#pragma unroll
    for (int tl = 0; tl < 4; ++tl) {
      const v16h bb = load_b_frag_pk(pk_wih, 1, wv * 64 + tl * 16);
      ac2[tl] = __builtin_amdgcn_wmma_f32_16x16x32_f16(
          false, a1f, false, bb, (short)0, ac2[tl], false, false);
    }
#pragma unroll
    for (int tl = 0; tl < 4; ++tl) acc[tl] = acc[tl] + ac2[tl];  // v_pk_add_f32
  }

  // layernorm over 512 gate cols per row
  float s[8], q[8];
#pragma unroll
  for (int r = 0; r < 8; ++r) {
    float ss = 0.f, qq = 0.f;
#pragma unroll
    for (int tl = 0; tl < 4; ++tl) { const float v = acc[tl][r]; ss += v; qq += v * v; }
    s[r] = red16(ss);
    q[r] = red16(qq);
  }
  if (n == 0) {
#pragma unroll
    for (int r = 0; r < 8; ++r) {
      red_s[wv * 16 + hw * 8 + r] = s[r];
      red_q[wv * 16 + hw * 8 + r] = q[r];
    }
  }
  __syncthreads();
  if (tid < 16) {
    float S = 0.f, Q = 0.f;
#pragma unroll
    for (int w2 = 0; w2 < 8; ++w2) { S += red_s[w2 * 16 + tid]; Q += red_q[w2 * 16 + tid]; }
    const float mu = S * (1.0f / GG);
    sm_mu[tid] = mu;
    sm_rs[tid] = rsqrtf(Q * (1.0f / GG) - mu * mu + EPSV);
  }
  __syncthreads();

#pragma unroll
  for (int r = 0; r < 8; ++r) {
    const int rl = hw * 8 + r;
    const float mu = sm_mu[rl], rs = sm_rs[rl];
#pragma unroll
    for (int tl = 0; tl < 4; ++tl) {
      const int col = wv * 64 + tl * 16 + n;
      const float v = (acc[tl][r] - mu) * rs * ln_i_w[col] + ln_i_b[col];
      ig[((size_t)t * BB + b0 + rl) * GG + col] = v;
    }
  }
}

// =====================================================================
// Phase 2: persistent LN-LSTM recurrence. grid = 16 WGs x 256 threads,
// WG owns 16 batch rows; W_hh B-fragments pinned in registers (from packed global).
// =====================================================================
__global__ void __launch_bounds__(256, 1)
lstm_kernel(const _Float16* __restrict__ pk_whh,
            const float* __restrict__ ln_h_w, const float* __restrict__ ln_h_b,
            const float* __restrict__ ln_c_w, const float* __restrict__ ln_c_b,
            const float* __restrict__ h0, const float* __restrict__ c0,
            const int* __restrict__ plen, const float* __restrict__ ig,
            float* __restrict__ out) {
  extern __shared__ char smem[];
  _Float16* lds_h = (_Float16*)smem;                 // 16x128 f16 = 4KB
  float* lds_g  = (float*)(smem + 4096);             // 16x512 f32 = 32KB
  float* s_lnhw = (float*)(smem + 4096 + 32768);     // 512
  float* s_lnhb = s_lnhw + 512;
  float* s_lncw = s_lnhb + 512;                      // 128
  float* s_lncb = s_lncw + 128;
  float* red_s  = s_lncb + 128;                      // 128
  float* red_q  = red_s + 128;
  float* sm_mu  = red_q + 128;                       // 16
  float* sm_rs  = sm_mu + 16;                        // 16

  const int tid = threadIdx.x;
  const int b0 = blockIdx.x << 4;
  const int wv = tid >> 5, hw = (tid >> 4) & 1, n = tid & 15;

  for (int i = tid; i < GG; i += 256) { s_lnhw[i] = ln_h_w[i]; s_lnhb[i] = ln_h_b[i]; }
  if (tid < HH) { s_lncw[tid] = ln_c_w[tid]; s_lncb[tid] = ln_c_b[tid]; }
  for (int i = tid; i < 16 * HH; i += 256)
    lds_h[i] = (_Float16)h0[(b0 + (i >> 7)) * HH + (i & 127)];

  // pin W_hh fragments in VGPRs: 4 N-tiles x 4 K-steps per wave (step-invariant)
  v16h bf[4][4];
#pragma unroll
  for (int tl = 0; tl < 4; ++tl)
#pragma unroll
    for (int kk = 0; kk < 4; ++kk)
      bf[tl][kk] = load_b_frag_pk(pk_whh, kk, wv * 64 + tl * 16);

  // per-thread cell state: row = tid>>4 (local batch row), cols j0..j0+7
  const int row = tid >> 4;
  const int j0 = (tid & 15) << 3;
  float c[8];
#pragma unroll
  for (int jj = 0; jj < 8; ++jj) c[jj] = c0[(b0 + row) * HH + j0 + jj];
  const int mylen = plen[b0 + row];
  const int gtype = wv >> 1;  // cols 64*wv..: 0=i, 1=f, 2=g, 3=o

  __syncthreads();

  for (int t = 0; t < LL; ++t) {
    // prefetch this step's ig slab (16x512 f32 = 256 lines of 128B)
    __builtin_prefetch((const char*)(ig + ((size_t)t * BB + b0) * GG) + tid * 128, 0, 0);

    // two independent accumulator chains (even/odd kk) -> dependent WMMAs 8 apart
    v8f acc[4], ac2[4];
#pragma unroll
    for (int tl = 0; tl < 4; ++tl)
#pragma unroll
      for (int r = 0; r < 8; ++r) { acc[tl][r] = 0.0f; ac2[tl][r] = 0.0f; }

#pragma unroll
    for (int kp = 0; kp < 2; ++kp) {
      const v16h ae = load_a_frag(lds_h, HH, (2 * kp) * 32);
      const v16h ao = load_a_frag(lds_h, HH, (2 * kp + 1) * 32);
#pragma unroll
      for (int tl = 0; tl < 4; ++tl)
        acc[tl] = __builtin_amdgcn_wmma_f32_16x16x32_f16(
            false, ae, false, bf[tl][2 * kp], (short)0, acc[tl], false, false);
#pragma unroll
      for (int tl = 0; tl < 4; ++tl)
        ac2[tl] = __builtin_amdgcn_wmma_f32_16x16x32_f16(
            false, ao, false, bf[tl][2 * kp + 1], (short)0, ac2[tl], false, false);
    }
#pragma unroll
    for (int tl = 0; tl < 4; ++tl) acc[tl] = acc[tl] + ac2[tl];  // v_pk_add_f32

    // LN over 512 gate cols per row (half-wave shuffles + cross-wave LDS)
    float s[8], q[8];
#pragma unroll
    for (int r = 0; r < 8; ++r) {
      float ss = 0.f, qq = 0.f;
#pragma unroll
      for (int tl = 0; tl < 4; ++tl) { const float v = acc[tl][r]; ss += v; qq += v * v; }
      s[r] = red16(ss);
      q[r] = red16(qq);
    }
    if (n == 0) {
#pragma unroll
      for (int r = 0; r < 8; ++r) {
        red_s[wv * 16 + hw * 8 + r] = s[r];
        red_q[wv * 16 + hw * 8 + r] = q[r];
      }
    }
    __syncthreads();
    if (tid < 16) {
      float S = 0.f, Q = 0.f;
#pragma unroll
      for (int w2 = 0; w2 < 8; ++w2) { S += red_s[w2 * 16 + tid]; Q += red_q[w2 * 16 + tid]; }
      const float mu = S * (1.0f / GG);
      sm_mu[tid] = mu;
      sm_rs[tid] = rsqrtf(Q * (1.0f / GG) - mu * mu + EPSV);
    }
    __syncthreads();

    // gates = act( LN(h@Whh^T)*w + b + ig )  -> staged in LDS
#pragma unroll
    for (int r = 0; r < 8; ++r) {
      const int rl = hw * 8 + r;
      const float mu = sm_mu[rl], rs = sm_rs[rl];
#pragma unroll
      for (int tl = 0; tl < 4; ++tl) {
        const int col = wv * 64 + tl * 16 + n;
        float v = (acc[tl][r] - mu) * rs * s_lnhw[col] + s_lnhb[col]
                  + ig[((size_t)t * BB + b0 + rl) * GG + col];
        v = (gtype == 2) ? tanh_fast(v) : sigm(v);
        lds_g[rl * GG + col] = v;
      }
    }
    __syncthreads();

    // c update + LN over 128 (pure intra-halfwave shuffles) + h
    float ss = 0.f, qq = 0.f;
#pragma unroll
    for (int jj = 0; jj < 8; ++jj) {
      const int j = j0 + jj;
      const float gi = lds_g[row * GG + j];
      const float gf = lds_g[row * GG + 128 + j];
      const float gg = lds_g[row * GG + 256 + j];
      const float cn = gf * c[jj] + gi * gg;
      c[jj] = cn;
      ss += cn; qq += cn * cn;
    }
    ss = red16(ss);
    qq = red16(qq);
    const float cmu = ss * (1.0f / HH);
    const float crs = rsqrtf(qq * (1.0f / HH) - cmu * cmu + EPSV);
    const bool emit = (t == mylen - 1);
#pragma unroll
    for (int jj = 0; jj < 8; ++jj) {
      const int j = j0 + jj;
      const float go = lds_g[row * GG + 384 + j];
      const float hval = go * tanh_fast((c[jj] - cmu) * crs * s_lncw[j] + s_lncb[j]);
      lds_h[row * HH + j] = (_Float16)hval;     // A operand for next step
      if (emit) out[(b0 + row) * HH + j] = hval;
    }
    __syncthreads();  // lds_h complete before next step's A-frag loads
  }
}

extern "C" void kernel_launch(void* const* d_in, const int* in_sizes, int n_in,
                              void* d_out, int out_size, void* d_ws, size_t ws_size,
                              hipStream_t stream) {
  (void)in_sizes; (void)n_in; (void)out_size; (void)ws_size;
  const float* peaks  = (const float*)d_in[0];
  const int*   plen   = (const int*)  d_in[1];
  const float* w_mz1  = (const float*)d_in[2];
  const float* b_mz1  = (const float*)d_in[3];
  const float* w_mz2  = (const float*)d_in[4];
  const float* b_mz2  = (const float*)d_in[5];
  const float* w_in1  = (const float*)d_in[6];
  const float* b_in1  = (const float*)d_in[7];
  const float* w_in2  = (const float*)d_in[8];
  const float* b_in2  = (const float*)d_in[9];
  const float* W_ih   = (const float*)d_in[10];
  const float* W_hh   = (const float*)d_in[11];
  const float* ln_i_w = (const float*)d_in[12];
  const float* ln_i_b = (const float*)d_in[13];
  const float* ln_h_w = (const float*)d_in[14];
  const float* ln_h_b = (const float*)d_in[15];
  const float* ln_c_w = (const float*)d_in[16];
  const float* ln_c_b = (const float*)d_in[17];
  const float* h0     = (const float*)d_in[18];
  const float* c0     = (const float*)d_in[19];

  // ws layout: packed W_ih f16 (64KB) | packed W_hh f16 (128KB) | ig_all f32 (256MB)
  _Float16* pk_wih = (_Float16*)d_ws;
  _Float16* pk_whh = (_Float16*)((char*)d_ws + 65536);
  float*    ig     = (float*)((char*)d_ws + 65536 + 131072);
  float*    out    = (float*)d_out;  // B*H f32

  pack_kernel<<<dim3(GG * F2 / 512), dim3(256), 0, stream>>>(W_ih, pk_wih, 6);
  pack_kernel<<<dim3(GG * HH / 512), dim3(256), 0, stream>>>(W_hh, pk_whh, 7);

  const size_t shm1 = 2048 + 512 + 512 + 64 + 64;                    // 3200
  ig_kernel<<<dim3(LL * (BB / 16)), dim3(256), shm1, stream>>>(
      peaks, plen, w_mz1, b_mz1, w_mz2, b_mz2, w_in1, b_in1, w_in2, b_in2,
      pk_wih, ln_i_w, ln_i_b, ig);

  const size_t shm2 = 4096 + 32768 + 2048 + 2048 + 512 + 512 + 512 + 512 + 64 + 64; // 43136
  lstm_kernel<<<dim3(BB / 16), dim3(256), shm2, stream>>>(
      pk_whh, ln_h_w, ln_h_b, ln_c_w, ln_c_b, h0, c0, plen, ig, out);
}
